// MedBERTTextExpert_17291538334410
// MI455X (gfx1250) — compile-verified, matrix operand order checked by reading, not run on hardware
//
#include <hip/hip_runtime.h>
#include <stdint.h>

// Problem constants (match reference)
#define BB 32
#define SS 128
#define TT 65536
#define DD 768
#define NSEG (BB * SS)          // 4096
#define LN_EPS 1e-5f

// GEMM tiling
#define BM 128                  // block M tile (8 waves x 16)
#define BN 64                   // block N tile
#define KSTEP 64                // K per LDS stage (2 WMMA substeps)
#define PAD_K 72                // padded K-stride in elements (144 B: 16B-aligned, bank-spread)

typedef __attribute__((ext_vector_type(8)))  __bf16 v8bf;
typedef __attribute__((ext_vector_type(16))) __bf16 v16bf;
typedef __attribute__((ext_vector_type(8)))  float  v8f;
typedef __attribute__((ext_vector_type(4)))  int    v4i;

#define GLOBAL_AS __attribute__((address_space(1)))
#define LDS_AS    __attribute__((address_space(3)))

#if defined(__gfx1250__) && \
    __has_builtin(__builtin_amdgcn_global_load_async_to_lds_b128) && \
    __has_builtin(__builtin_amdgcn_s_wait_asynccnt)
#define HAVE_ASYNC_LDS 1
#else
#define HAVE_ASYNC_LDS 0
#endif

__device__ __forceinline__ unsigned short f32_to_bf16(float f) {
    unsigned int u = __float_as_uint(f);
    unsigned int r = (u + 0x7FFFu + ((u >> 16) & 1u)) >> 16;   // round-to-nearest-even
    return (unsigned short)r;
}

__device__ __forceinline__ int lower_bound(const int* __restrict__ a, int n, int v) {
    int lo = 0, hi = n;
    while (lo < hi) {
        int m = (lo + hi) >> 1;
        if (a[m] < v) lo = m + 1; else hi = m;
    }
    return lo;
}

// 16-byte global -> LDS copy; async DMA (ASYNCcnt) when the toolchain has it.
__device__ __forceinline__ void cp16_g2l(const void* gsrc, void* lds_dst) {
#if HAVE_ASYNC_LDS
    __builtin_amdgcn_global_load_async_to_lds_b128(
        (GLOBAL_AS v4i*)(uintptr_t)gsrc,
        (LDS_AS v4i*)(uint32_t)(uintptr_t)lds_dst, 0, 0);
#else
    *(v8bf*)lds_dst = *(const v8bf*)gsrc;
#endif
}

__device__ __forceinline__ void stage_fence() {
#if HAVE_ASYNC_LDS
    __builtin_amdgcn_s_wait_asynccnt(0);
#endif
    __syncthreads();
}

// ---------------------------------------------------------------------------
// Kernel 1: ragged segment mean-pool + 5 embedding adds + LayerNorm -> bf16
// One block (256 threads, 8 waves) per sentence slot. 768 = 3 dims/thread.
// ---------------------------------------------------------------------------
__global__ void __launch_bounds__(256)
pool_enrich_ln_kernel(const int* __restrict__ token_ids,
                      const int* __restrict__ segment_ids,
                      const int* __restrict__ sec_ids,
                      const int* __restrict__ temp_ids,
                      const int* __restrict__ neg_ids,
                      const int* __restrict__ pos_ids,
                      const int* __restrict__ ts_ids,
                      const float* __restrict__ tok_tab,
                      const float* __restrict__ sec_tab,
                      const float* __restrict__ temp_tab,
                      const float* __restrict__ neg_tab,
                      const float* __restrict__ pos_tab,
                      const float* __restrict__ ts_tab,
                      const float* __restrict__ gamma,
                      const float* __restrict__ beta,
                      unsigned short* __restrict__ normed_bf16)
{
    const int seg = blockIdx.x;
    const int tid = threadIdx.x;

    __shared__ int   s_lo, s_hi;
    __shared__ float red[256];
    __shared__ float s_mu, s_rstd;

    if (tid == 0) {
        s_lo = lower_bound(segment_ids, TT, seg);
        s_hi = lower_bound(segment_ids, TT, seg + 1);
    }
    __syncthreads();
    const int lo = s_lo, hi = s_hi;

    const int d0 = tid, d1 = tid + 256, d2 = tid + 512;

    // Gather-accumulate token rows (coalesced: whole block reads one row)
    float a0 = 0.f, a1 = 0.f, a2 = 0.f;
    for (int t = lo; t < hi; ++t) {
        const float* rowp = tok_tab + (size_t)token_ids[t] * DD;
        if (t + 1 < hi)  // hint next row into cache (global_prefetch_b8)
            __builtin_prefetch(tok_tab + (size_t)token_ids[t + 1] * DD + d0, 0, 0);
        a0 += rowp[d0];
        a1 += rowp[d1];
        a2 += rowp[d2];
    }
    const float inv = 1.0f / fmaxf((float)(hi - lo), 1.0f);

    const int se = sec_ids[seg], te = temp_ids[seg], ne = neg_ids[seg];
    const int po = pos_ids[seg], ts = ts_ids[seg];

    float x0 = a0 * inv + sec_tab[se * DD + d0] + temp_tab[te * DD + d0]
             + neg_tab[ne * DD + d0] + pos_tab[po * DD + d0] + ts_tab[ts * DD + d0];
    float x1 = a1 * inv + sec_tab[se * DD + d1] + temp_tab[te * DD + d1]
             + neg_tab[ne * DD + d1] + pos_tab[po * DD + d1] + ts_tab[ts * DD + d1];
    float x2 = a2 * inv + sec_tab[se * DD + d2] + temp_tab[te * DD + d2]
             + neg_tab[ne * DD + d2] + pos_tab[po * DD + d2] + ts_tab[ts * DD + d2];

    // mean
    red[tid] = x0 + x1 + x2;
    __syncthreads();
    for (int s = 128; s > 0; s >>= 1) {
        if (tid < s) red[tid] += red[tid + s];
        __syncthreads();
    }
    if (tid == 0) s_mu = red[0] * (1.0f / (float)DD);
    __syncthreads();
    const float mu = s_mu;

    // variance
    const float e0 = x0 - mu, e1 = x1 - mu, e2 = x2 - mu;
    red[tid] = e0 * e0 + e1 * e1 + e2 * e2;
    __syncthreads();
    for (int s = 128; s > 0; s >>= 1) {
        if (tid < s) red[tid] += red[tid + s];
        __syncthreads();
    }
    if (tid == 0) s_rstd = rsqrtf(red[0] * (1.0f / (float)DD) + LN_EPS);
    __syncthreads();
    const float rs = s_rstd;

    unsigned short* outp = normed_bf16 + (size_t)seg * DD;
    outp[d0] = f32_to_bf16(e0 * rs * gamma[d0] + beta[d0]);
    outp[d1] = f32_to_bf16(e1 * rs * gamma[d1] + beta[d1]);
    outp[d2] = f32_to_bf16(e2 * rs * gamma[d2] + beta[d2]);
}

// ---------------------------------------------------------------------------
// Kernel 2: proj_w f32 [e][d] -> bf16 [e][d] (K-contiguous for B fragments)
// ---------------------------------------------------------------------------
__global__ void __launch_bounds__(256)
convert_w_kernel(const float* __restrict__ w, unsigned short* __restrict__ wb)
{
    int i = blockIdx.x * 256 + threadIdx.x;   // 0 .. 768*768-1 (exact grid)
    wb[i] = f32_to_bf16(w[i]);
}

// ---------------------------------------------------------------------------
// Kernel 3: GEMM  out[m,e] = sum_d normed[m,d] * W[e,d] + bias[e]
// M=4096, N=768, K=768. bf16 WMMA 16x16x32, f32 accumulate.
// Double-buffered LDS stages filled with GLOBAL_LOAD_ASYNC_TO_LDS_B128
// (ASYNCcnt) when available; fragments come from LDS via ds_load_b128.
// Block = 256 thr = 8 waves; wave tile 16(M) x 64(N); grid (32,12) exact.
// ---------------------------------------------------------------------------
__global__ void __launch_bounds__(256)
gemm_wmma_kernel(const unsigned short* __restrict__ a_bf16,
                 const unsigned short* __restrict__ w_bf16,
                 const float* __restrict__ bias,
                 float* __restrict__ out)
{
    const __bf16* A = (const __bf16*)a_bf16;   // [4096][768] row-major
    const __bf16* W = (const __bf16*)w_bf16;   // [768][768]  (e-major, d contiguous)

    __shared__ __align__(16) unsigned short Asm[2][BM * PAD_K];  // 2 x 18 KB
    __shared__ __align__(16) unsigned short Bsm[2][BN * PAD_K];  // 2 x  9 KB

    const int tid  = threadIdx.x;
    const int wave = tid >> 5;
    const int lane = tid & 31;
    const int row  = lane & 15;   // M row (A frag) / N col (B frag) within tile
    const int kh   = lane >> 4;   // K-chunk selector per ISA 16-bit layout

    const int mblk = blockIdx.x * BM;
    const int n0   = blockIdx.y * BN;

    // ---- stage fill: A tile BM x KSTEP (1024 x 16B chunks), B tile BN x KSTEP (512)
    auto stage = [&](int buf, int k0) {
#pragma unroll
        for (int i = 0; i < 4; ++i) {            // A: 4 chunks / thread
            int c    = tid + i * 256;
            int rowl = c >> 3;                   // 0..127
            int koff = (c & 7) * 8;              // elements
            cp16_g2l(A + (size_t)(mblk + rowl) * DD + k0 + koff,
                     &Asm[buf][rowl * PAD_K + koff]);
        }
#pragma unroll
        for (int i = 0; i < 2; ++i) {            // B: 2 chunks / thread
            int c    = tid + i * 256;
            int rowl = c >> 3;                   // 0..63
            int koff = (c & 7) * 8;
            cp16_g2l(W + (size_t)(n0 + rowl) * DD + k0 + koff,
                     &Bsm[buf][rowl * PAD_K + koff]);
        }
    };

    v8f acc[4] = {};

    stage(0, 0);
    stage_fence();

    int buf = 0;
    for (int k0 = 0; k0 < DD; k0 += KSTEP) {
        if (k0 + KSTEP < DD) stage(buf ^ 1, k0 + KSTEP);   // prefetch next stage

        const unsigned short* As = &Asm[buf][(wave * 16 + row) * PAD_K + kh * 8];
#pragma unroll
        for (int kk = 0; kk < 2; ++kk) {         // two 16x16x32 substeps per stage
            // A fragment: lane<16 holds K {0..7,16..23}; lane>=16 holds {8..15,24..31}
            v16bf afrag;
            {
                const unsigned short* ap = As + kk * 32;
                ((v8bf*)&afrag)[0] = *(const v8bf*)(ap);
                ((v8bf*)&afrag)[1] = *(const v8bf*)(ap + 16);
            }
#pragma unroll
            for (int j = 0; j < 4; ++j) {
                v16bf bfrag;
                const unsigned short* bp =
                    &Bsm[buf][(j * 16 + row) * PAD_K + kk * 32 + kh * 8];
                ((v8bf*)&bfrag)[0] = *(const v8bf*)(bp);
                ((v8bf*)&bfrag)[1] = *(const v8bf*)(bp + 16);

                acc[j] = __builtin_amdgcn_wmma_f32_16x16x32_bf16(
                    /*neg_a=*/false, afrag, /*neg_b=*/false, bfrag,
                    /*c_mod=*/(short)0, acc[j], /*reuse_a=*/false, /*reuse_b=*/false);
            }
        }
        stage_fence();                            // wait our async fills + barrier
        buf ^= 1;
    }

    // D layout: VGPR r -> M = r (lanes 0-15) / r+8 (lanes 16-31); N = lane%16
    const int m0 = mblk + wave * 16;
#pragma unroll
    for (int j = 0; j < 4; ++j) {
        const int ocol = n0 + j * 16 + row;
        const float b = bias[ocol];
#pragma unroll
        for (int r = 0; r < 8; ++r) {
            const int orow = m0 + kh * 8 + r;
            out[(size_t)orow * DD + ocol] = acc[j][r] + b;
        }
    }
}

// ---------------------------------------------------------------------------
// Kernel 4: padding_mask (all false) -> zero the tail of d_out
// ---------------------------------------------------------------------------
__global__ void zero_tail_kernel(float* __restrict__ out, int n)
{
    int i = blockIdx.x * 256 + threadIdx.x;
    if (i < n) out[(size_t)NSEG * DD + i] = 0.0f;
}

// ---------------------------------------------------------------------------
extern "C" void kernel_launch(void* const* d_in, const int* in_sizes, int n_in,
                              void* d_out, int out_size, void* d_ws, size_t ws_size,
                              hipStream_t stream)
{
    const int*   token_ids   = (const int*)d_in[0];
    const int*   segment_ids = (const int*)d_in[1];
    const int*   sec_ids     = (const int*)d_in[2];
    const int*   temp_ids    = (const int*)d_in[3];
    const int*   neg_ids     = (const int*)d_in[4];
    const int*   pos_ids     = (const int*)d_in[5];
    const int*   ts_ids      = (const int*)d_in[6];
    const float* tok_tab     = (const float*)d_in[7];
    const float* sec_tab     = (const float*)d_in[8];
    const float* temp_tab    = (const float*)d_in[9];
    const float* neg_tab     = (const float*)d_in[10];
    const float* pos_tab     = (const float*)d_in[11];
    const float* ts_tab      = (const float*)d_in[12];
    const float* ln_gamma    = (const float*)d_in[13];
    const float* ln_beta     = (const float*)d_in[14];
    const float* proj_w      = (const float*)d_in[15];
    const float* proj_b      = (const float*)d_in[16];

    // Workspace: normed bf16 [4096*768], W bf16 [768*768]  (~7.5 MB total)
    unsigned short* nrm = (unsigned short*)d_ws;
    unsigned short* wb  = nrm + (size_t)NSEG * DD;

    pool_enrich_ln_kernel<<<NSEG, 256, 0, stream>>>(
        token_ids, segment_ids, sec_ids, temp_ids, neg_ids, pos_ids, ts_ids,
        tok_tab, sec_tab, temp_tab, neg_tab, pos_tab, ts_tab,
        ln_gamma, ln_beta, nrm);

    convert_w_kernel<<<(DD * DD) / 256, 256, 0, stream>>>(proj_w, wb);

    gemm_wmma_kernel<<<dim3(NSEG / BM, DD / BN), 256, 0, stream>>>(
        nrm, wb, proj_b, (float*)d_out);

    const int tail = out_size - NSEG * DD;   // padding_mask elements
    if (tail > 0)
        zero_tail_kernel<<<(tail + 255) / 256, 256, 0, stream>>>((float*)d_out, tail);
}
